// TransformerBlock_40080634807019
// MI455X (gfx1250) — matrix operand order repaired and checked by reading, not compiled, not run on hardware
//
#include <hip/hip_runtime.h>
#include <hip/hip_bf16.h>

// ---------------------------------------------------------------------------
// Problem constants (match reference)
// ---------------------------------------------------------------------------
constexpr int D_M = 1024;   // d_model
constexpr int NH  = 16;     // heads
constexpr int DK  = 64;     // head dim
constexpr int FF  = 4096;   // d_ff
constexpr int SQ  = 2048;   // seq
constexpr int BA  = 2;      // batch
constexpr int NT  = BA * SQ;// total tokens = 4096

typedef __bf16 bf16_t;
typedef __attribute__((ext_vector_type(16))) __bf16 v16bf;
typedef __attribute__((ext_vector_type(8)))  __bf16 v8bf;
typedef __attribute__((ext_vector_type(8)))  float  v8f;
typedef __attribute__((ext_vector_type(4)))  unsigned int v4u;
typedef __attribute__((ext_vector_type(8)))  int v8i;
typedef __attribute__((ext_vector_type(4)))  int v4i;

#if defined(__has_builtin)
#if __has_builtin(__builtin_amdgcn_tensor_load_to_lds) && \
    __has_builtin(__builtin_amdgcn_s_wait_tensorcnt)
#define USE_TDM 1
#endif
#endif
#ifndef USE_TDM
#define USE_TDM 0
#endif

// ---------------------------------------------------------------------------
// WMMA fragment helpers (CDNA5 16-bit layouts, wave32)
//
// A 16x32 (MxK), row-major source [rows, ld]:
//   lane L: row = row0 + (L&15); K-chunks [k0 + (L>>4)*8 .. +7] and
//   [k0+16+(L>>4)*8 .. +7]  (two contiguous 16B loads)
// B 32x16 (KxN) from row-major W[N, K] (B[k][n] = W[n][k]):
//   lane L: col = col0 + (L&15); K-chunks [k0 + (L>>4)*8 .. +7] and
//   [k0+16+(L>>4)*8 .. +7]  (two contiguous 16B loads)
// C/D 16x16 f32: VGPR v, lane L -> row v + 8*(L>>4), col (L&15)
// ---------------------------------------------------------------------------
__device__ __forceinline__ v16bf load_a_frag(const bf16_t* base, int ld,
                                             int row0, int k0) {
  const int lane = threadIdx.x & 31;
  const int half = lane >> 4;
  const bf16_t* p = base + (size_t)(row0 + (lane & 15)) * ld + k0 + half * 8;
  v8bf lo = *(const v8bf*)p;
  v8bf hi = *(const v8bf*)(p + 16);
  v16bf r;
#pragma unroll
  for (int i = 0; i < 8; ++i) { r[i] = lo[i]; r[i + 8] = hi[i]; }
  return r;
}

__device__ __forceinline__ v16bf load_b_frag(const bf16_t* base, int ld,
                                             int col0, int k0) {
  const int lane = threadIdx.x & 31;
  const int half = lane >> 4;
  const bf16_t* p = base + (size_t)(col0 + (lane & 15)) * ld + k0 + half * 8;
  v8bf lo = *(const v8bf*)p;
  v8bf hi = *(const v8bf*)(p + 16);
  v16bf r;
#pragma unroll
  for (int i = 0; i < 8; ++i) { r[i] = lo[i]; r[i + 8] = hi[i]; }
  return r;
}

__device__ __forceinline__ v8f wmma_bf16(v16bf a, v16bf b, v8f c) {
  // (neg_a, A, neg_b, B, c_mod, C, reuse_a, reuse_b)
  return __builtin_amdgcn_wmma_f32_16x16x32_bf16(false, a, false, b,
                                                 (short)0, c, false, false);
}

#if USE_TDM
// ---------------------------------------------------------------------------
// Tensor Data Mover: DMA a 64-row x 32-col bf16 tile (row stride = K elements)
// from global into LDS (packed 64x32 row-major).  D# per CDNA5 ISA:
//   group0: count=1 | lds_addr | global_addr[56:0] | type=2
//   group1: data_size=2B | tensor_dim0=K | tensor_dim1=M | tile_dim0=32 |
//           tile_dim1=64 | tensor_dim0_stride=K
// This toolchain exposes the 6-arg builtin:
//   (uint32x4 g0, int32x8 g1, int32x4, int32x4, int32x8, i32 cpol)
// trailing groups are zero (2D tensor, no gather/iterate).
// ---------------------------------------------------------------------------
__device__ __forceinline__ void tdm_load_a_tile(const bf16_t* gsrc,
                                                unsigned int lds_byte_off,
                                                int K, int M) {
  const unsigned long long ga = (unsigned long long)(uintptr_t)gsrc;
  v4u g0;
  g0[0] = 1u;                                    // count=1, no gather
  g0[1] = lds_byte_off;                          // lds_addr
  g0[2] = (unsigned int)ga;                      // global_addr[31:0]
  g0[3] = (unsigned int)((ga >> 32) & 0x01FFFFFFull) | (2u << 30);  // type=2
  v8i g1;
  g1[0] = (int)(1u << 16);                       // data_size = 2 bytes
  g1[1] = (int)(((unsigned)K & 0xFFFFu) << 16);  // tensor_dim0[15:0]
  g1[2] = (int)(((unsigned)K >> 16) | (((unsigned)M & 0xFFFFu) << 16));
  g1[3] = (int)(((unsigned)M >> 16) | (32u << 16));  // tile_dim0 = 32
  g1[4] = 64;                                    // tile_dim1 = 64
  g1[5] = K;                                     // tensor_dim0_stride lo
  g1[6] = 0;
  g1[7] = 0;
  const v4i z4 = {0, 0, 0, 0};                   // groups 2/3 unused (2D)
  const v8i z8 = {0, 0, 0, 0, 0, 0, 0, 0};
  __builtin_amdgcn_tensor_load_to_lds(g0, g1, z4, z4, z8, 0);
}
#endif

// ---------------------------------------------------------------------------
// f32 -> bf16 cast (weights)
// ---------------------------------------------------------------------------
__global__ __launch_bounds__(256) void cvt_bf16_kernel(
    const float* __restrict__ in, bf16_t* __restrict__ out, int n) {
  int i = blockIdx.x * 256 + threadIdx.x;
  if (i < n) out[i] = (bf16_t)in[i];
}

// ---------------------------------------------------------------------------
// RMSNorm: one block (8 waves) per token row; output bf16
// ---------------------------------------------------------------------------
__global__ __launch_bounds__(256) void rmsnorm_kernel(
    const float* __restrict__ x, const float* __restrict__ w,
    bf16_t* __restrict__ out) {
  const int row = blockIdx.x;
  const float* xr = x + (size_t)row * D_M;
  float ss = 0.f;
  for (int i = threadIdx.x; i < D_M; i += 256) { float v = xr[i]; ss += v * v; }
#pragma unroll
  for (int off = 16; off >= 1; off >>= 1) ss += __shfl_xor(ss, off, 32);
  __shared__ float red[8];
  __shared__ float s_rinv;
  if ((threadIdx.x & 31) == 0) red[threadIdx.x >> 5] = ss;
  __syncthreads();
  if (threadIdx.x == 0) {
    float t = 0.f;
#pragma unroll
    for (int i = 0; i < 8; ++i) t += red[i];
    s_rinv = rsqrtf(t / (float)D_M + 1e-5f);
  }
  __syncthreads();
  const float rinv = s_rinv;
  bf16_t* outr = out + (size_t)row * D_M;
  for (int i = threadIdx.x; i < D_M; i += 256)
    outr[i] = (bf16_t)(xr[i] * rinv * w[i]);
}

// ---------------------------------------------------------------------------
// Generic bf16 WMMA GEMM:  out[M,Nc] = A[M,K] * W[Nc,K]^T  (+ epilogue)
// Block = 8 waves (2 x 4).  Wave tile = (MI*16) x (NI*16); block M-tile = 64.
// A-tile staged in LDS by the Tensor Data Mover (double buffered); B streams
// direct from global (contiguous-K fragments).
// EPI: 0 = f32 store, 1 = f32 store + residual, 2 = SwiGLU->bf16 (uses W1),
//      3 = bf16 store
// ---------------------------------------------------------------------------
template <int MI, int NI, int EPI>
__global__ __launch_bounds__(256) void gemm_bf16_kernel(
    const bf16_t* __restrict__ A, const bf16_t* __restrict__ W0,
    const bf16_t* __restrict__ W1, const float* __restrict__ Res,
    float* __restrict__ outF, bf16_t* __restrict__ outB,
    int M, int Nc, int K) {
  const int wid  = threadIdx.x >> 5;
  const int lane = threadIdx.x & 31;
  const int half = lane >> 4;
  const int nl   = lane & 15;
  const int rowBlk = blockIdx.y * 64;                       // block M origin
  const int waveM  = wid >> 2;                              // 0..1
  const int row0   = rowBlk + waveM * (MI * 16);
  const int col0 = blockIdx.x * (4 * NI * 16) + (wid & 3) * (NI * 16);

  v8f acc0[MI][NI];
  v8f acc1[MI][NI];
#pragma unroll
  for (int m = 0; m < MI; ++m)
#pragma unroll
    for (int n = 0; n < NI; ++n) {
      acc0[m][n] = {};
      if (EPI == 2) acc1[m][n] = {};
    }

#if USE_TDM
  __shared__ alignas(16) bf16_t atile[2][64 * 32];          // 2 x 4KB
  const unsigned int abase = (unsigned int)(uintptr_t)&atile[0][0];
  // preload first A tile
  if (wid == 0) {
    tdm_load_a_tile(A + (size_t)rowBlk * K, abase, K, M);
    __builtin_amdgcn_s_wait_tensorcnt((short)0);
  }
  __syncthreads();
  int buf = 0;
#endif

  for (int k0 = 0; k0 < K; k0 += 32) {
#if USE_TDM
    if (k0 + 32 < K && wid == 0)  // DMA next A tile while computing this one
      tdm_load_a_tile(A + (size_t)rowBlk * K + (k0 + 32),
                      abase + (unsigned)((buf ^ 1) * 64 * 32 * 2), K, M);
    v16bf a[MI];
#pragma unroll
    for (int m = 0; m < MI; ++m)
      a[m] = load_a_frag(&atile[buf][0], 32, waveM * (MI * 16) + 16 * m, 0);
#else
    v16bf a[MI];
#pragma unroll
    for (int m = 0; m < MI; ++m) a[m] = load_a_frag(A, K, row0 + 16 * m, k0);
#endif
    if (k0 + 64 < K)  // global_prefetch_b8 of next B K-tile
      __builtin_prefetch(W0 + (size_t)col0 * K + k0 + 64, 0, 1);
#pragma unroll
    for (int n = 0; n < NI; ++n) {
      v16bf b0 = load_b_frag(W0, K, col0 + 16 * n, k0);
#pragma unroll
      for (int m = 0; m < MI; ++m) acc0[m][n] = wmma_bf16(a[m], b0, acc0[m][n]);
      if (EPI == 2) {
        v16bf b1 = load_b_frag(W1, K, col0 + 16 * n, k0);
#pragma unroll
        for (int m = 0; m < MI; ++m) acc1[m][n] = wmma_bf16(a[m], b1, acc1[m][n]);
      }
    }
#if USE_TDM
    if (k0 + 32 < K) {
      if (wid == 0) __builtin_amdgcn_s_wait_tensorcnt((short)0);
      __syncthreads();
      buf ^= 1;
    }
#endif
  }

#pragma unroll
  for (int m = 0; m < MI; ++m)
#pragma unroll
    for (int n = 0; n < NI; ++n) {
      const int c = col0 + 16 * n + nl;
#pragma unroll
      for (int v = 0; v < 8; ++v) {
        const int r = row0 + 16 * m + v + 8 * half;
        const size_t idx = (size_t)r * Nc + c;
        const float x1 = acc0[m][n][v];
        if (EPI == 0) {
          outF[idx] = x1;
        } else if (EPI == 1) {
          outF[idx] = Res[idx] + x1;
        } else if (EPI == 3) {
          outB[idx] = (bf16_t)x1;
        } else {  // SwiGLU: x1 * sigmoid(x1) * x3
          const float x3 = acc1[m][n][v];
          outB[idx] = (bf16_t)(x1 / (1.0f + __expf(-x1)) * x3);
        }
      }
    }
}

// ---------------------------------------------------------------------------
// RoPE + relayout.  In: q/k/v bf16 [B,S,H*DK].  Out: q,k bf16 [B,H,S,DK];
// v transposed bf16 [B,H,DK,S] (so P*V B-fragments are contiguous).
// ---------------------------------------------------------------------------
__global__ __launch_bounds__(256) void rope_kernel(
    const bf16_t* __restrict__ q, const bf16_t* __restrict__ k,
    const bf16_t* __restrict__ v, const int* __restrict__ pos,
    bf16_t* __restrict__ qo, bf16_t* __restrict__ ko,
    bf16_t* __restrict__ vt) {
  int t = blockIdx.x * 256 + threadIdx.x;
  if (t >= BA * SQ * NH * (DK / 2)) return;
  const int i = t & 31;  t >>= 5;   // rotary pair index
  const int h = t & 15;  t >>= 4;
  const int s = t & (SQ - 1); t >>= 11;
  const int b = t;
  const float inv = __expf(-(float)(2 * i) * (logf(10000.0f) / (float)DK));
  const float ang = (float)pos[s] * inv;
  const float cs = __cosf(ang), sn = __sinf(ang);
  const size_t in_idx  = ((size_t)(b * SQ + s)) * D_M + h * DK + 2 * i;
  const size_t out_idx = (((size_t)(b * NH + h)) * SQ + s) * DK + 2 * i;
  {
    const float xe = (float)q[in_idx], xo = (float)q[in_idx + 1];
    qo[out_idx]     = (bf16_t)(xe * cs - xo * sn);
    qo[out_idx + 1] = (bf16_t)(xe * sn + xo * cs);
  }
  {
    const float xe = (float)k[in_idx], xo = (float)k[in_idx + 1];
    ko[out_idx]     = (bf16_t)(xe * cs - xo * sn);
    ko[out_idx + 1] = (bf16_t)(xe * sn + xo * cs);
  }
  const size_t vt_idx = (((size_t)(b * NH + h)) * DK + 2 * i) * SQ + s;
  vt[vt_idx]      = v[in_idx];
  vt[vt_idx + SQ] = v[in_idx + 1];
}

// ---------------------------------------------------------------------------
// Flash attention (causal).  Grid: (SQ/128, B*H). 8 waves; each wave owns a
// 16-row Q tile, streams 32-wide K blocks with online softmax; P bounces
// through a per-wave LDS tile to switch C-layout -> A-layout.  O -> bf16
// [B,S,H*DK] ready for the W_o GEMM.
// ---------------------------------------------------------------------------
__global__ __launch_bounds__(256) void attn_kernel(
    const bf16_t* __restrict__ Q, const bf16_t* __restrict__ K,
    const bf16_t* __restrict__ Vt, bf16_t* __restrict__ O) {
  const int bh = blockIdx.y;
  const int b = bh >> 4;
  const int h = bh & 15;
  const int wid  = threadIdx.x >> 5;
  const int lane = threadIdx.x & 31;
  const int half = lane >> 4;
  const int nl   = lane & 15;
  const int qRow0 = blockIdx.x * 128 + wid * 16;
  const bf16_t* Qp = Q  + (size_t)bh * SQ * DK;
  const bf16_t* Kp = K  + (size_t)bh * SQ * DK;
  const bf16_t* Vp = Vt + (size_t)bh * DK * SQ;

  const v16bf qa0 = load_a_frag(Qp, DK, qRow0, 0);
  const v16bf qa1 = load_a_frag(Qp, DK, qRow0, 32);

  v8f oacc[4];
#pragma unroll
  for (int t = 0; t < 4; ++t) oacc[t] = {};
  float mrow[8], lrow[8];
#pragma unroll
  for (int v = 0; v < 8; ++v) { mrow[v] = -1e30f; lrow[v] = 0.f; }

  __shared__ alignas(16) bf16_t pshm[8][16 * 32];  // per-wave P staging
  bf16_t* pw = &pshm[wid][0];

  const float scale = 0.125f;  // 1/sqrt(64)
  const int kEnd = qRow0 + 16;
  for (int kb = 0; kb < kEnd; kb += 32) {
    // scores: Q(16x64) x K^T(64x32) -> two 16x16 f32 tiles
    v8f sc0 = {}, sc1 = {};
    {
      v16bf bk = load_b_frag(Kp, DK, kb, 0);       sc0 = wmma_bf16(qa0, bk, sc0);
      bk      = load_b_frag(Kp, DK, kb, 32);       sc0 = wmma_bf16(qa1, bk, sc0);
      bk      = load_b_frag(Kp, DK, kb + 16, 0);   sc1 = wmma_bf16(qa0, bk, sc1);
      bk      = load_b_frag(Kp, DK, kb + 16, 32);  sc1 = wmma_bf16(qa1, bk, sc1);
    }
#pragma unroll
    for (int v = 0; v < 8; ++v) {
      const int r = qRow0 + v + 8 * half;          // this lane's row for VGPR v
      float s0 = sc0[v] * scale;
      float s1 = sc1[v] * scale;
      if (kb + nl > r)      s0 = -1e30f;           // causal mask
      if (kb + 16 + nl > r) s1 = -1e30f;
      float mx = fmaxf(s0, s1);
#pragma unroll
      for (int off = 8; off >= 1; off >>= 1) mx = fmaxf(mx, __shfl_xor(mx, off, 16));
      const float mnew = fmaxf(mrow[v], mx);
      const float ef = __expf(mrow[v] - mnew);
      const float p0 = __expf(s0 - mnew);
      const float p1 = __expf(s1 - mnew);
      float rs = p0 + p1;
#pragma unroll
      for (int off = 8; off >= 1; off >>= 1) rs += __shfl_xor(rs, off, 16);
      lrow[v] = lrow[v] * ef + rs;
      mrow[v] = mnew;
#pragma unroll
      for (int t = 0; t < 4; ++t) oacc[t][v] *= ef;
      // stage P (C-layout scatter; same-wave DS ops are in-order)
      pw[(v + 8 * half) * 32 + nl]      = (bf16_t)p0;
      pw[(v + 8 * half) * 32 + 16 + nl] = (bf16_t)p1;
    }
    // reload P in A-layout, accumulate O += P(16x32) x V(32x64)
    const v16bf pa = load_a_frag(pw, 32, 0, 0);
#pragma unroll
    for (int t = 0; t < 4; ++t) {
      v16bf bv = load_b_frag(Vp, SQ, t * 16, kb);  // B[k][n] = V[kb+k, t*16+n]
      oacc[t] = wmma_bf16(pa, bv, oacc[t]);
    }
  }

#pragma unroll
  for (int t = 0; t < 4; ++t) {
    const int dcol = h * DK + t * 16 + nl;
#pragma unroll
    for (int v = 0; v < 8; ++v) {
      const int r = qRow0 + v + 8 * half;
      O[((size_t)(b * SQ + r)) * D_M + dcol] = (bf16_t)(oacc[t][v] / lrow[v]);
    }
  }
}

// ---------------------------------------------------------------------------
// Host driver
// ---------------------------------------------------------------------------
extern "C" void kernel_launch(void* const* d_in, const int* in_sizes, int n_in,
                              void* d_out, int out_size, void* d_ws,
                              size_t ws_size, hipStream_t stream) {
  const float* x   = (const float*)d_in[0];
  const int*   pos = (const int*)d_in[1];
  const float* wq  = (const float*)d_in[2];
  const float* wk  = (const float*)d_in[3];
  const float* wv  = (const float*)d_in[4];
  const float* wo  = (const float*)d_in[5];
  const float* ln1 = (const float*)d_in[6];
  const float* ln2 = (const float*)d_in[7];
  const float* w1  = (const float*)d_in[8];
  const float* w2  = (const float*)d_in[9];
  const float* w3  = (const float*)d_in[10];

  char* p = (char*)d_ws;
  auto carve = [&](size_t bytes) -> void* {
    void* r = (void*)p;
    p += (bytes + 255) & ~(size_t)255;
    return r;
  };
  const size_t ND = (size_t)NT * D_M;
  bf16_t* xn1   = (bf16_t*)carve(ND * 2);
  bf16_t* wqb   = (bf16_t*)carve((size_t)D_M * D_M * 2);
  bf16_t* wkb   = (bf16_t*)carve((size_t)D_M * D_M * 2);
  bf16_t* wvb   = (bf16_t*)carve((size_t)D_M * D_M * 2);
  bf16_t* wob   = (bf16_t*)carve((size_t)D_M * D_M * 2);
  bf16_t* w1b   = (bf16_t*)carve((size_t)FF * D_M * 2);
  bf16_t* w3b   = (bf16_t*)carve((size_t)FF * D_M * 2);
  bf16_t* w2b   = (bf16_t*)carve((size_t)D_M * FF * 2);
  bf16_t* qt    = (bf16_t*)carve(ND * 2);
  bf16_t* kt    = (bf16_t*)carve(ND * 2);
  bf16_t* vt0   = (bf16_t*)carve(ND * 2);
  bf16_t* qr    = (bf16_t*)carve(ND * 2);
  bf16_t* kr    = (bf16_t*)carve(ND * 2);
  bf16_t* vtr   = (bf16_t*)carve(ND * 2);
  bf16_t* ob    = (bf16_t*)carve(ND * 2);
  float*  xmid  = (float*)carve(ND * 4);
  bf16_t* xn2   = (bf16_t*)carve(ND * 2);
  bf16_t* gated = (bf16_t*)carve((size_t)NT * FF * 2);

  const dim3 blk(256);
  auto cvt = [&](const float* src, bf16_t* dst, size_t n) {
    cvt_bf16_kernel<<<dim3((unsigned)((n + 255) / 256)), blk, 0, stream>>>(
        src, dst, (int)n);
  };
  cvt(wq, wqb, (size_t)D_M * D_M);
  cvt(wk, wkb, (size_t)D_M * D_M);
  cvt(wv, wvb, (size_t)D_M * D_M);
  cvt(wo, wob, (size_t)D_M * D_M);
  cvt(w1, w1b, (size_t)FF * D_M);
  cvt(w3, w3b, (size_t)FF * D_M);
  cvt(w2, w2b, (size_t)D_M * FF);

  // 1) RMSNorm(x) -> xn1 (bf16)
  rmsnorm_kernel<<<dim3(NT), blk, 0, stream>>>(x, ln1, xn1);

  // 2) Q/K/V projections (bf16 WMMA, bf16 out)
  const dim3 gq(D_M / 256, NT / 64);
  gemm_bf16_kernel<2, 4, 3><<<gq, blk, 0, stream>>>(
      xn1, wqb, nullptr, nullptr, nullptr, qt, NT, D_M, D_M);
  gemm_bf16_kernel<2, 4, 3><<<gq, blk, 0, stream>>>(
      xn1, wkb, nullptr, nullptr, nullptr, kt, NT, D_M, D_M);
  gemm_bf16_kernel<2, 4, 3><<<gq, blk, 0, stream>>>(
      xn1, wvb, nullptr, nullptr, nullptr, vt0, NT, D_M, D_M);

  // 3) RoPE on q,k; transpose v
  rope_kernel<<<dim3((BA * SQ * NH * 32 + 255) / 256), blk, 0, stream>>>(
      qt, kt, vt0, pos, qr, kr, vtr);

  // 4) Causal flash attention -> ob (bf16 [B,S,D])
  attn_kernel<<<dim3(SQ / 128, BA * NH), blk, 0, stream>>>(qr, kr, vtr, ob);

  // 5) x_mid = x + ob @ wo^T   (f32 + residual)
  gemm_bf16_kernel<2, 4, 1><<<gq, blk, 0, stream>>>(
      ob, wob, nullptr, x, xmid, nullptr, NT, D_M, D_M);

  // 6) RMSNorm(x_mid) -> xn2
  rmsnorm_kernel<<<dim3(NT), blk, 0, stream>>>(xmid, ln2, xn2);

  // 7) Fused SwiGLU: gated = silu(xn2@w1^T) * (xn2@w3^T)  (bf16)
  gemm_bf16_kernel<2, 2, 2><<<dim3(FF / 128, NT / 64), blk, 0, stream>>>(
      xn2, w1b, w3b, nullptr, nullptr, gated, NT, FF, D_M);

  // 8) out = x_mid + gated @ w2^T  (f32 final output)
  gemm_bf16_kernel<2, 4, 1><<<dim3(D_M / 256, NT / 64), blk, 0, stream>>>(
      gated, w2b, nullptr, xmid, (float*)d_out, nullptr, NT, D_M, FF);
}
